// TransformerBlock_63290638074502
// MI455X (gfx1250) — matrix operand order repaired and checked by reading, not compiled, not run on hardware
//
#include <hip/hip_runtime.h>

// ---------------- types ----------------
typedef __bf16 bf16_t;
typedef __attribute__((ext_vector_type(16))) __bf16 v16bf;
typedef __attribute__((ext_vector_type(8)))  __bf16 v8bf;
typedef __attribute__((ext_vector_type(2)))  __bf16 v2bf;
typedef __attribute__((ext_vector_type(8)))  float  v8f;

// ---------------- problem constants ----------------
constexpr int Btok = 16, Hh = 64, Ww = 64, Ee = 512, Dd = 64, Ff = 2048;
constexpr int NT = Btok * Hh * Ww;   // 65536 tokens

// ---------------- helpers ----------------
static __device__ __forceinline__ __bf16 f2bf(float f) {
    return (__bf16)f;                 // compiler picks native v_cvt when available
}

static __device__ __forceinline__ v2bf pk_bf16(float lo, float hi) {
#if __has_builtin(__builtin_amdgcn_cvt_pk_bf16_f32)
    return __builtin_amdgcn_cvt_pk_bf16_f32(lo, hi);
#else
    v2bf r;
    r[0] = (__bf16)lo;
    r[1] = (__bf16)hi;
    return r;
#endif
}

static __device__ __forceinline__ v8f vzero8() {
    v8f z;
#pragma unroll
    for (int i = 0; i < 8; ++i) z[i] = 0.0f;
    return z;
}

static __device__ __forceinline__ v8f wmma_bf16(v16bf a, v16bf b, v8f c) {
    // D = A(16x32 bf16) * B(32x16 bf16) + C(16x16 f32)
    return __builtin_amdgcn_wmma_f32_16x16x32_bf16(false, a, false, b, (short)0, c,
                                                   false, false);
}

// A fragment (16x32, row-major source, f32 elements -> bf16 via packed cvt)
static __device__ __forceinline__ v16bf load_a_f32(const float* __restrict__ A, int lda,
                                                   int row0, int k0, int lane) {
    int r  = row0 + (lane & 15);
    int kb = k0 + ((lane >> 4) << 3);          // +0 or +8
    const float* p = A + (size_t)r * lda + kb;
    v8f c0 = *(const v8f*)p;                   // K = kb .. kb+7
    v8f c1 = *(const v8f*)(p + 16);            // K = kb+16 .. kb+23
    v16bf f;
#pragma unroll
    for (int i = 0; i < 4; ++i) {
        v2bf p0 = pk_bf16(c0[2 * i], c0[2 * i + 1]);
        v2bf p1 = pk_bf16(c1[2 * i], c1[2 * i + 1]);
        f[2 * i]     = p0[0];
        f[2 * i + 1] = p0[1];
        f[8 + 2 * i]     = p1[0];
        f[8 + 2 * i + 1] = p1[1];
    }
    return f;
}

// A fragment (16x32, row-major source, bf16 elements)
static __device__ __forceinline__ v16bf load_a_bf16(const bf16_t* __restrict__ A, int lda,
                                                    int row0, int k0, int lane) {
    int r  = row0 + (lane & 15);
    int kb = k0 + ((lane >> 4) << 3);
    const bf16_t* p = A + (size_t)r * lda + kb;
    v8bf c0 = *(const v8bf*)p;
    v8bf c1 = *(const v8bf*)(p + 16);
    v16bf f;
#pragma unroll
    for (int i = 0; i < 8; ++i) { f[i] = c0[i]; f[8 + i] = c1[i]; }
    return f;
}

// B fragment (32x16) from a column-major (N x K, "BT") bf16 matrix:
// lane holds column n0+(lane&15), K values kb..kb+7 and kb+16..kb+23, kb = k0+(lane>>4)*8
static __device__ __forceinline__ v16bf load_b(const bf16_t* __restrict__ BT, int ldb,
                                               int col0, int k0, int lane) {
    int c  = col0 + (lane & 15);
    int kb = k0 + ((lane >> 4) << 3);
    const bf16_t* p = BT + (size_t)c * ldb + kb;
    v8bf c0 = *(const v8bf*)p;
    v8bf c1 = *(const v8bf*)(p + 16);
    v16bf f;
#pragma unroll
    for (int i = 0; i < 8; ++i) { f[i] = c0[i]; f[8 + i] = c1[i]; }
    return f;
}

// C/D element (g, lane) -> (row0 + (lane>=16)*8 + g, col0 + lane%16)

// ---------------- kernel 0: weight pack (f32 KxN row-major -> bf16 NxK) --------
__global__ void pack_wt_kernel(const float* __restrict__ Wsrc, bf16_t* __restrict__ Wt,
                               int K, int N) {
    int idx = blockIdx.x * blockDim.x + threadIdx.x;
    if (idx >= K * N) return;
    int n = idx / K, k = idx - n * K;
    Wt[(size_t)n * K + k] = f2bf(Wsrc[(size_t)k * N + n]);
}

// ---------------- kernel 1: fused QKV projection -------------------------------
// C = x(65536x512) @ Wcat(512x640); epilogue: bias + rel pos, scatter q/k/vT.
// N-split is block-uniform: bx==0 -> q, bx==1 -> k, bx>=2 -> v.
__global__ __launch_bounds__(256) void qkv_kernel(
    const float* __restrict__ x, const bf16_t* __restrict__ WcatT,
    const float* __restrict__ bq, const float* __restrict__ bk,
    const float* __restrict__ bv, const float* __restrict__ rel_h,
    const float* __restrict__ rel_w, bf16_t* __restrict__ qb,
    bf16_t* __restrict__ kb, bf16_t* __restrict__ vT) {
    const int lane = threadIdx.x & 31, wv = threadIdx.x >> 5;
    const int bx = blockIdx.x;
    const int n0 = bx * 64;
    const int m0 = blockIdx.y * 64;
    const int mt  = (2 * wv) >> 2;       // wave's M tile (0..3)
    const int nta = (2 * wv) & 3;        // wave's first N tile
    v8f acc0 = vzero8(), acc1 = vzero8();
    for (int k0 = 0; k0 < Ee; k0 += 32) {
        v16bf a  = load_a_f32(x, Ee, m0 + mt * 16, k0, lane);
        v16bf b0 = load_b(WcatT, Ee, n0 + nta * 16, k0, lane);
        v16bf b1 = load_b(WcatT, Ee, n0 + (nta + 1) * 16, k0, lane);
        acc0 = wmma_bf16(a, b0, acc0);
        acc1 = wmma_bf16(a, b1, acc1);
    }
#pragma unroll
    for (int t = 0; t < 2; ++t) {
        v8f acc = t ? acc1 : acc0;
        int nt = nta + t;
#pragma unroll
        for (int g = 0; g < 8; ++g) {
            int m = m0 + mt * 16 + ((lane >> 4) << 3) + g;
            int n = n0 + nt * 16 + (lane & 15);
            float val = acc[g];
            int bh = m >> 6, w = m & 63, h = bh & 63;
            if (bx == 0) {                                  // q + bq + rel_h[h]
                val += bq[n] + rel_h[h * Dd + n];
                qb[(size_t)m * Dd + n] = f2bf(val);
            } else if (bx == 1) {                           // k + bk + rel_w[w]
                int c = n - 64;
                val += bk[c] + rel_w[w * Dd + c];
                kb[(size_t)m * Dd + c] = f2bf(val);
            } else {                                        // v + bv, transposed
                int c = n - 128;
                val += bv[c];
                vT[((size_t)bh * Ee + c) * 64 + w] = f2bf(val);
            }
        }
    }
}

// ---------------- kernel 2: axial attention + residual + LN1 -------------------
// one block per (b,h) row; S=q@k^T (WMMA), softmax, O=P@V (WMMA), LN -> h1
__global__ __launch_bounds__(256) void attn_ln1_kernel(
    const bf16_t* __restrict__ qb, const bf16_t* __restrict__ kb,
    const bf16_t* __restrict__ vT, const float* __restrict__ x,
    const float* __restrict__ g1, const float* __restrict__ be1,
    float* __restrict__ h1) {
    extern __shared__ char smem[];
    float*  sS = (float*)smem;                                   // 64x64 f32
    bf16_t* sP = (bf16_t*)(smem + 64 * 64 * 4);                  // 64x64 bf16
    float*  sO = (float*)(smem + 64 * 64 * 4 + 64 * 64 * 2);     // 64x512 f32
    float*  sSum = (float*)((char*)sO + 64 * 512 * 4);           // 64
    float*  sSq  = sSum + 64;                                    // 64

    const int tid = threadIdx.x, lane = tid & 31, wv = tid >> 5;
    const int bh = blockIdx.x;
    const bf16_t* q  = qb + (size_t)bh * 64 * Dd;
    const bf16_t* kk = kb + (size_t)bh * 64 * Dd;
    const bf16_t* vt = vT + (size_t)bh * Ee * 64;

    // ---- S = q @ k^T  (M=N=64, K=64); k row-major == B column-major ----
    {
        const int mt = (2 * wv) >> 2, nta = (2 * wv) & 3;
        v8f acc0 = vzero8(), acc1 = vzero8();
#pragma unroll
        for (int k0 = 0; k0 < Dd; k0 += 32) {
            v16bf a  = load_a_bf16(q, Dd, mt * 16, k0, lane);
            v16bf b0 = load_b(kk, Dd, nta * 16, k0, lane);
            v16bf b1 = load_b(kk, Dd, (nta + 1) * 16, k0, lane);
            acc0 = wmma_bf16(a, b0, acc0);
            acc1 = wmma_bf16(a, b1, acc1);
        }
#pragma unroll
        for (int t = 0; t < 2; ++t) {
            v8f acc = t ? acc1 : acc0;
            int nt = nta + t;
#pragma unroll
            for (int g = 0; g < 8; ++g) {
                int i = mt * 16 + ((lane >> 4) << 3) + g;
                int j = nt * 16 + (lane & 15);
                sS[i * 64 + j] = acc[g];
            }
        }
    }
    __syncthreads();

    // ---- softmax rows (no 1/sqrt(d) scale, matching reference) ----
    if (tid < 64) {
        float mx = -1e30f;
        for (int j = 0; j < 64; ++j) mx = fmaxf(mx, sS[tid * 64 + j]);
        float s = 0.0f;
        for (int j = 0; j < 64; ++j) {
            float e = __expf(sS[tid * 64 + j] - mx);
            sS[tid * 64 + j] = e;
            s += e;
        }
        float inv = 1.0f / s;
        for (int j = 0; j < 32; ++j) {
            v2bf p = pk_bf16(sS[tid * 64 + 2 * j] * inv, sS[tid * 64 + 2 * j + 1] * inv);
            *(v2bf*)&sP[tid * 64 + 2 * j] = p;
        }
    }
    __syncthreads();

    // ---- O = P @ V  (M=64, N=512, K=64); vT is column-major B ----
    {
        const int mt = wv >> 1;
        const int ntbase = (wv & 1) * 16;
        v16bf a0 = load_a_bf16(sP, 64, mt * 16, 0, lane);
        v16bf a1 = load_a_bf16(sP, 64, mt * 16, 32, lane);
#pragma unroll 4
        for (int j = 0; j < 16; ++j) {
            int c0 = (ntbase + j) * 16;
            v16bf bb0 = load_b(vt, 64, c0, 0, lane);
            v16bf bb1 = load_b(vt, 64, c0, 32, lane);
            v8f acc = vzero8();
            acc = wmma_bf16(a0, bb0, acc);
            acc = wmma_bf16(a1, bb1, acc);
#pragma unroll
            for (int g = 0; g < 8; ++g) {
                int i = mt * 16 + ((lane >> 4) << 3) + g;
                sO[i * 512 + c0 + (lane & 15)] = acc[g];
            }
        }
    }
    if (tid < 64) { sSum[tid] = 0.0f; sSq[tid] = 0.0f; }
    __syncthreads();

    // ---- residual + LN statistics (4 threads per row) ----
    {
        const int r = tid & 63, seg = tid >> 6;       // 4 segments of 128 channels
        const float* xr = x + ((size_t)bh * 64 + r) * Ee;
        float ps = 0.0f, pq = 0.0f;
        for (int c = seg * 128; c < seg * 128 + 128; ++c) {
            float val = xr[c] + sO[r * 512 + c];
            sO[r * 512 + c] = val;
            ps += val;
            pq += val * val;
        }
        atomicAdd(&sSum[r], ps);
        atomicAdd(&sSq[r], pq);
    }
    __syncthreads();
    {
        const int r = tid & 63, seg = tid >> 6;
        float mean = sSum[r] * (1.0f / 512.0f);
        float var  = sSq[r] * (1.0f / 512.0f) - mean * mean;
        float rstd = rsqrtf(var + 1e-5f);
        float* outp = h1 + ((size_t)bh * 64 + r) * Ee;
        for (int c = seg * 128; c < seg * 128 + 128; ++c)
            outp[c] = (sO[r * 512 + c] - mean) * rstd * g1[c] + be1[c];
    }
}

// ---------------- kernel 3: FFN up-projection + exact GELU ---------------------
__global__ __launch_bounds__(256) void ffn1_kernel(
    const float* __restrict__ h1, const bf16_t* __restrict__ W1T,
    const float* __restrict__ b1, bf16_t* __restrict__ tb) {
    const int lane = threadIdx.x & 31, wv = threadIdx.x >> 5;
    const int n0 = blockIdx.x * 64;
    const int m0 = blockIdx.y * 64;
    const int mt = (2 * wv) >> 2, nta = (2 * wv) & 3;
    v8f acc0 = vzero8(), acc1 = vzero8();
    for (int k0 = 0; k0 < Ee; k0 += 32) {
        if (k0 + 32 < Ee) {
            const float* pf =
                h1 + (size_t)(m0 + mt * 16 + (lane & 15)) * Ee + k0 + 32;
            __builtin_prefetch(pf, 0, 3);
        }
        v16bf a   = load_a_f32(h1, Ee, m0 + mt * 16, k0, lane);
        v16bf b0  = load_b(W1T, Ee, n0 + nta * 16, k0, lane);
        v16bf b1v = load_b(W1T, Ee, n0 + (nta + 1) * 16, k0, lane);
        acc0 = wmma_bf16(a, b0, acc0);
        acc1 = wmma_bf16(a, b1v, acc1);
    }
#pragma unroll
    for (int t = 0; t < 2; ++t) {
        v8f acc = t ? acc1 : acc0;
        int nt = nta + t;
#pragma unroll
        for (int g = 0; g < 8; ++g) {
            int m = m0 + mt * 16 + ((lane >> 4) << 3) + g;
            int n = n0 + nt * 16 + (lane & 15);
            float z = acc[g] + b1[n];
            float ge = 0.5f * z * (1.0f + erff(z * 0.70710678118654752f));
            tb[(size_t)m * Ff + n] = f2bf(ge);
        }
    }
}

// ---------------- kernel 4: FFN down-projection + residual + LN2 ---------------
// block = 32 rows x all 512 cols so LN2 fuses; io holds h1 on entry, out on exit
__global__ __launch_bounds__(256) void ffn2_ln2_kernel(
    const bf16_t* __restrict__ tb, const bf16_t* __restrict__ W2T,
    const float* __restrict__ b2, const float* __restrict__ g2,
    const float* __restrict__ be2, float* __restrict__ io) {
    extern __shared__ char smem[];
    float* sAcc = (float*)smem;          // 32x512 f32
    float* sSum = sAcc + 32 * 512;       // 32
    float* sSq  = sSum + 32;             // 32

    const int tid = threadIdx.x, lane = tid & 31, wv = tid >> 5;
    const int m0 = blockIdx.x * 32;
    const int mt = wv & 1;               // 2 M tiles
    const int ntbase = (wv >> 1) * 8;    // 8 N tiles per wave (of 32)

    v8f acc[8];
#pragma unroll
    for (int j = 0; j < 8; ++j) acc[j] = vzero8();

    for (int k0 = 0; k0 < Ff; k0 += 32) {
        if (k0 + 32 < Ff) {
            const bf16_t* pf =
                tb + (size_t)(m0 + mt * 16 + (lane & 15)) * Ff + k0 + 32;
            __builtin_prefetch(pf, 0, 3);
        }
        v16bf a = load_a_bf16(tb, Ff, m0 + mt * 16, k0, lane);
#pragma unroll
        for (int j = 0; j < 8; ++j) {
            v16bf bb = load_b(W2T, Ff, (ntbase + j) * 16, k0, lane);
            acc[j] = wmma_bf16(a, bb, acc[j]);   // A fragment reused 8x
        }
    }
    // bias + residual (read h1 from io before any thread overwrites it)
#pragma unroll
    for (int j = 0; j < 8; ++j) {
#pragma unroll
        for (int g = 0; g < 8; ++g) {
            int r = mt * 16 + ((lane >> 4) << 3) + g;   // 0..31
            int n = (ntbase + j) * 16 + (lane & 15);
            sAcc[r * 512 + n] = acc[j][g] + b2[n] + io[(size_t)(m0 + r) * Ee + n];
        }
    }
    if (tid < 32) { sSum[tid] = 0.0f; sSq[tid] = 0.0f; }
    __syncthreads();

    const int r = tid & 31, seg = tid >> 5;             // 8 segments of 64 channels
    float ps = 0.0f, pq = 0.0f;
    for (int c = seg * 64; c < seg * 64 + 64; ++c) {
        float v = sAcc[r * 512 + c];
        ps += v;
        pq += v * v;
    }
    atomicAdd(&sSum[r], ps);
    atomicAdd(&sSq[r], pq);
    __syncthreads();

    float mean = sSum[r] * (1.0f / 512.0f);
    float var  = sSq[r] * (1.0f / 512.0f) - mean * mean;
    float rstd = rsqrtf(var + 1e-5f);
    float* outp = io + (size_t)(m0 + r) * Ee;
    for (int c = seg * 64; c < seg * 64 + 64; ++c)
        outp[c] = (sAcc[r * 512 + c] - mean) * rstd * g2[c] + be2[c];
}

// ---------------- host launcher ----------------
extern "C" void kernel_launch(void* const* d_in, const int* in_sizes, int n_in,
                              void* d_out, int out_size, void* d_ws, size_t ws_size,
                              hipStream_t stream) {
    const float* x     = (const float*)d_in[0];
    const float* Wq    = (const float*)d_in[1];
    const float* bq    = (const float*)d_in[2];
    const float* Wk    = (const float*)d_in[3];
    const float* bk    = (const float*)d_in[4];
    const float* Wv    = (const float*)d_in[5];
    const float* bv    = (const float*)d_in[6];
    const float* rel_h = (const float*)d_in[7];
    const float* rel_w = (const float*)d_in[8];
    const float* ln1_g = (const float*)d_in[9];
    const float* ln1_b = (const float*)d_in[10];
    const float* W1    = (const float*)d_in[11];
    const float* b1    = (const float*)d_in[12];
    const float* W2    = (const float*)d_in[13];
    const float* b2    = (const float*)d_in[14];
    const float* ln2_g = (const float*)d_in[15];
    const float* ln2_b = (const float*)d_in[16];
    float* out = (float*)d_out;   // also used as h1 scratch between kernels 2..4

    char* ws = (char*)d_ws;
    size_t off = 0;
    auto carve = [&](size_t bytes) -> char* {
        char* p = ws + off;
        off += (bytes + 255) & ~(size_t)255;
        return p;
    };
    bf16_t* WcatT = (bf16_t*)carve((size_t)640 * 512 * 2);    // [q|k|v]^T
    bf16_t* W1T   = (bf16_t*)carve((size_t)2048 * 512 * 2);
    bf16_t* W2T   = (bf16_t*)carve((size_t)512 * 2048 * 2);
    bf16_t* qb    = (bf16_t*)carve((size_t)NT * 64 * 2);
    bf16_t* kb    = (bf16_t*)carve((size_t)NT * 64 * 2);
    bf16_t* vT    = (bf16_t*)carve((size_t)NT * 512 * 2);
    bf16_t* tb    = (bf16_t*)carve((size_t)NT * 2048 * 2);

    // 0) pack weights (bf16, column-major = WMMA-B friendly)
    pack_wt_kernel<<<(512 * 64 + 255) / 256, 256, 0, stream>>>(Wq, WcatT, 512, 64);
    pack_wt_kernel<<<(512 * 64 + 255) / 256, 256, 0, stream>>>(Wk, WcatT + (size_t)64 * 512, 512, 64);
    pack_wt_kernel<<<(512 * 512 + 255) / 256, 256, 0, stream>>>(Wv, WcatT + (size_t)128 * 512, 512, 512);
    pack_wt_kernel<<<(512 * 2048 + 255) / 256, 256, 0, stream>>>(W1, W1T, 512, 2048);
    pack_wt_kernel<<<(2048 * 512 + 255) / 256, 256, 0, stream>>>(W2, W2T, 2048, 512);

    // 1) fused QKV projection
    qkv_kernel<<<dim3(10, 1024), 256, 0, stream>>>(x, WcatT, bq, bk, bv, rel_h, rel_w,
                                                   qb, kb, vT);

    // 2) axial attention + residual + LN1  (h1 -> d_out)
    size_t lds2 = (size_t)64 * 64 * 4 + 64 * 64 * 2 + 64 * 512 * 4 + 2 * 64 * 4;
    attn_ln1_kernel<<<1024, 256, lds2, stream>>>(qb, kb, vT, x, ln1_g, ln1_b, out);

    // 3) FFN up + GELU
    ffn1_kernel<<<dim3(32, 1024), 256, 0, stream>>>(out, W1T, b1, tb);

    // 4) FFN down + residual + LN2 (reads h1 from d_out, overwrites with result)
    size_t lds4 = (size_t)32 * 512 * 4 + 2 * 32 * 4;
    ffn2_ln2_kernel<<<2048, 256, lds4, stream>>>(tb, W2T, b2, ln2_g, ln2_b, out);
}
